// AttentionalPooler_1864015806683
// MI455X (gfx1250) — matrix operand order repaired and checked by reading, compiled, not run on hardware
//
#include <hip/hip_runtime.h>
#include <hip/hip_bf16.h>
#include <math.h>

typedef unsigned short u16;
typedef __attribute__((ext_vector_type(16))) __bf16 v16bf;
typedef __attribute__((ext_vector_type(8)))  float  v8f;

union Frag { v16bf v; uint4 q[2]; };

__device__ __forceinline__ u16 f32_to_bf16(float f) {
    unsigned int u = __float_as_uint(f);
    unsigned int r = u + 0x7FFFu + ((u >> 16) & 1u);   // round-to-nearest-even
    return (u16)(r >> 16);
}

// ---------------------------------------------------------------------------
// Async global->LDS (CDNA5 ASYNCcnt path) with compile-safe fallback.
// Builtin signature (from hipcc diagnostic): (int4 AS1*, int4 AS3*, Imm, Imm)
// ---------------------------------------------------------------------------
#if __has_builtin(__builtin_amdgcn_global_load_async_to_lds_b128) && \
    __has_builtin(__builtin_amdgcn_s_wait_asynccnt)
#define USE_ASYNC_LDS 1
#else
#define USE_ASYNC_LDS 0
#endif

typedef int vi4_t __attribute__((vector_size(16)));

// Copy 16 bytes (8 bf16) global -> LDS.
__device__ __forceinline__ void copy16(const u16* __restrict__ g, u16* l) {
#if USE_ASYNC_LDS
    typedef __attribute__((address_space(1))) vi4_t* gp_t;
    typedef __attribute__((address_space(3))) vi4_t* lp_t;
    __builtin_amdgcn_global_load_async_to_lds_b128(
        (gp_t)(unsigned long long)(const void*)g,
        (lp_t)(unsigned int)(unsigned long long)(void*)l,
        0, 0);
#else
    *(uint4*)l = *(const uint4*)g;
#endif
}

__device__ __forceinline__ void wait_copies() {
#if USE_ASYNC_LDS
    __builtin_amdgcn_s_wait_asynccnt(0);
#endif
}

// ---------------------------------------------------------------------------
// f32 -> bf16 elementwise conversion
// ---------------------------------------------------------------------------
__global__ __launch_bounds__(256)
void cvt_f32_bf16_kernel(const float* __restrict__ x, u16* __restrict__ y, int n) {
    int i = blockIdx.x * 256 + threadIdx.x;
    if (i < n) y[i] = f32_to_bf16(x[i]);
}

// f32 [K][N] -> bf16 transposed [N][K]  (one-time weight prep)
__global__ __launch_bounds__(256)
void cvt_f32_bf16_T_kernel(const float* __restrict__ x, u16* __restrict__ y,
                           int K, int N) {
    int i = blockIdx.x * 256 + threadIdx.x;
    if (i < K * N) {
        int k = i / N, n = i - k * N;
        y[(size_t)n * K + k] = f32_to_bf16(x[i]);
    }
}

// ---------------------------------------------------------------------------
// Row LayerNorm -> bf16.  One block (256 thr = 8 waves) per row of length L.
// ---------------------------------------------------------------------------
__global__ __launch_bounds__(256)
void ln_bf16_kernel(const float* __restrict__ x, const float* __restrict__ g,
                    const float* __restrict__ bta, u16* __restrict__ y, int L) {
    __shared__ float red[16];
    const int row = blockIdx.x;
    const int tid = threadIdx.x;
    const int lane = tid & 31, wave = tid >> 5;
    const size_t base = (size_t)row * L;

    float s = 0.f, s2 = 0.f;
    for (int i = tid; i < L; i += 256) {
        float v = x[base + i];
        s += v; s2 += v * v;
    }
    for (int off = 16; off >= 1; off >>= 1) {
        s  += __shfl_xor(s,  off);
        s2 += __shfl_xor(s2, off);
    }
    if (lane == 0) { red[wave] = s; red[8 + wave] = s2; }
    __syncthreads();
    float ts = 0.f, ts2 = 0.f;
    #pragma unroll
    for (int w = 0; w < 8; ++w) { ts += red[w]; ts2 += red[8 + w]; }
    const float invL = 1.0f / (float)L;
    const float mean = ts * invL;
    const float var  = ts2 * invL - mean * mean;
    const float rstd = rsqrtf(var + 1e-5f);

    for (int i = tid; i < L; i += 256) {
        float v = (x[base + i] - mean) * rstd * g[i] + bta[i];
        y[base + i] = f32_to_bf16(v);
    }
}

// ---------------------------------------------------------------------------
// bf16 GEMM, fp32 accumulate: C = alpha * A[M,K] @ Bt[N,K]^T
//   A  : bf16 [M][K] row-major
//   Bt : bf16 [N][K] (pre-transposed B) -> LDS fills are straight copies
// 128x128 block tile, K-step 32, double-buffered LDS, 8 waves (4x2),
// wave tile 32x64.  Requires M%128==0, N%128==0, K%32==0 (true here).
// OMODE: 0 = fp32 C, 1 = bf16 C, 2 = KV split (K half -> Ck rows [m][768],
//        V half -> Cvt transposed [b*768 + (n-768)][4096], m = b*4096+nctx)
// ---------------------------------------------------------------------------
#define GS 40   // LDS row stride (u16); 80B = multiple of 16B

template<int OMODE>
__global__ __launch_bounds__(256)
void gemm_bf16_128x128(const u16* __restrict__ A, const u16* __restrict__ Bt,
                       void* __restrict__ Cvoid, u16* __restrict__ Cvt,
                       int M, int N, int K, float alpha) {
    __shared__ u16 As[2][128 * GS];
    __shared__ u16 Bs[2][128 * GS];

    const int tid  = threadIdx.x;
    const int lane = tid & 31, wave = tid >> 5;
    const int wm = wave & 3, wn = wave >> 2;          // 4 waves in M, 2 in N
    const int m0 = blockIdx.y * 128, n0 = blockIdx.x * 128;
    const int l16 = lane & 15;

    v8f acc[2][4];
    const v8f vzero = {0.f, 0.f, 0.f, 0.f, 0.f, 0.f, 0.f, 0.f};
    #pragma unroll
    for (int mt = 0; mt < 2; ++mt)
        #pragma unroll
        for (int nt = 0; nt < 4; ++nt) acc[mt][nt] = vzero;

    const int trow  = tid >> 1;             // 0..127 tile row (A: m, B: n)
    const int thalf = (tid & 1) * 16;       // 0/16 K offset
    const int akb   = (lane < 16) ? 0 : 8;  // A-frag K base
    const int bkb   = (lane < 16) ? 0 : 8;  // B-frag K base

    const u16* aptr = A  + (size_t)(m0 + trow) * K + thalf;
    const u16* bptr = Bt + (size_t)(n0 + trow) * K + thalf;
    u16* asl = &As[0][trow * GS + thalf];   // buf stride added below
    u16* bsl = &Bs[0][trow * GS + thalf];

    // prologue: fill buffer 0
    copy16(aptr,     asl);
    copy16(aptr + 8, asl + 8);
    copy16(bptr,     bsl);
    copy16(bptr + 8, bsl + 8);
    wait_copies();
    __syncthreads();

    int buf = 0;
    for (int k0 = 0; k0 < K; k0 += 32) {
        // stream next tile into the other buffer while computing this one
        if (k0 + 32 < K) {
            const int nb = buf ^ 1;
            copy16(aptr + k0 + 32,     asl + nb * 128 * GS);
            copy16(aptr + k0 + 40,     asl + nb * 128 * GS + 8);
            copy16(bptr + k0 + 32,     bsl + nb * 128 * GS);
            copy16(bptr + k0 + 40,     bsl + nb * 128 * GS + 8);
        }

        Frag a[2];
        #pragma unroll
        for (int mt = 0; mt < 2; ++mt) {
            const int row = wm * 32 + mt * 16 + l16;
            a[mt].q[0] = *(const uint4*)&As[buf][row * GS + akb];
            a[mt].q[1] = *(const uint4*)&As[buf][row * GS + akb + 16];
        }
        #pragma unroll
        for (int nt = 0; nt < 4; ++nt) {
            Frag b;
            const int col = wn * 64 + nt * 16 + l16;
            b.q[0] = *(const uint4*)&Bs[buf][col * GS + bkb];
            b.q[1] = *(const uint4*)&Bs[buf][col * GS + bkb + 16];
            #pragma unroll
            for (int mt = 0; mt < 2; ++mt)
                acc[mt][nt] = __builtin_amdgcn_wmma_f32_16x16x32_bf16(
                    false, a[mt].v, false, b.v, (short)0, acc[mt][nt], false, false);
        }

        wait_copies();       // this wave's next-tile copies landed
        __syncthreads();     // all waves done with 'buf' and copies visible
        buf ^= 1;
    }

    // epilogue: D layout — lane l, vgpr r -> m = base + (l>>4)*8 + r, n = base + (l&15)
    const int mb = m0 + wm * 32 + ((lane >> 4) << 3);
    const int nb = n0 + wn * 64 + l16;
    #pragma unroll
    for (int mt = 0; mt < 2; ++mt)
        #pragma unroll
        for (int nt = 0; nt < 4; ++nt)
            #pragma unroll
            for (int r = 0; r < 8; ++r) {
                const int m = mb + mt * 16 + r;
                const int n = nb + nt * 16;
                const float vv = acc[mt][nt][r] * alpha;
                if (OMODE == 0) {
                    ((float*)Cvoid)[(size_t)m * N + n] = vv;
                } else if (OMODE == 1) {
                    ((u16*)Cvoid)[(size_t)m * N + n] = f32_to_bf16(vv);
                } else {
                    const u16 bv = f32_to_bf16(vv);
                    if (n < 768) {      // K half: rows [m][768]
                        ((u16*)Cvoid)[(size_t)m * 768 + n] = bv;
                    } else {            // V half: transposed [b*768+(n-768)][4096]
                        const int bb = m >> 12, nctx = m & 4095;
                        Cvt[((size_t)bb * 768 + (n - 768)) * 4096 + nctx] = bv;
                    }
                }
            }
}

// ---------------------------------------------------------------------------
// Flash attention: one block per (h, b); 8 waves x 32 queries; dh = 96.
// qb : [256][768] bf16 (pre-scaled)
// kb_: [8*4096][768] bf16 K rows
// vtb: [8*768][4096] bf16 V transposed (d-major) -> straight-copy LDS fills
// ob : [8*256][768] bf16
// ---------------------------------------------------------------------------
#define KTS 104   // 208B stride
#define VTS 72    // 144B stride
#define PSS 72

__global__ __launch_bounds__(256)
void flash_attn_kernel(const u16* __restrict__ qb, const u16* __restrict__ kbuf,
                       const u16* __restrict__ vtb, u16* __restrict__ ob) {
    __shared__ u16 Kt[64 * KTS];      // K tile rows == B^T rows for Q.K^T
    __shared__ u16 Vt[96 * VTS];      // V tile, d-major: Vt[d][n]
    __shared__ u16 Ps[8][32 * PSS];   // per-wave P staging (D-layout -> A-layout)

    const int h = blockIdx.x, b = blockIdx.y;
    const int tid = threadIdx.x, lane = tid & 31, wave = tid >> 5;
    const int l16 = lane & 15, lhi = lane >> 4;
    const int akb = (lane < 16) ? 0 : 8;
    const int bkb = (lane < 16) ? 0 : 8;

    // preload Q fragments straight from global (A-fragment layout)
    Frag qf[2][3];
    #pragma unroll
    for (int mt = 0; mt < 2; ++mt)
        #pragma unroll
        for (int kk = 0; kk < 3; ++kk) {
            const u16* p = qb + (size_t)(wave * 32 + mt * 16 + l16) * 768
                              + h * 96 + kk * 32 + akb;
            qf[mt][kk].q[0] = *(const uint4*)p;
            qf[mt][kk].q[1] = *(const uint4*)(p + 16);
        }

    v8f oacc[2][6];
    const v8f vzero = {0.f, 0.f, 0.f, 0.f, 0.f, 0.f, 0.f, 0.f};
    float mrun[2][8], lrun[2][8];
    #pragma unroll
    for (int mt = 0; mt < 2; ++mt) {
        #pragma unroll
        for (int dt = 0; dt < 6; ++dt) oacc[mt][dt] = vzero;
        #pragma unroll
        for (int r = 0; r < 8; ++r) { mrun[mt][r] = -1e30f; lrun[mt][r] = 0.f; }
    }

    for (int nc = 0; nc < 4096; nc += 64) {
        __syncthreads();              // all waves done with previous tile
        // K tile: 64 rows x 96 el = 768 x 16B chunks, 3 per thread
        #pragma unroll
        for (int j = 0; j < 3; ++j) {
            const int c = tid + j * 256;      // 0..767
            const int r = c / 12, cs = (c % 12) * 8;
            copy16(kbuf + (size_t)(b * 4096 + nc + r) * 768 + h * 96 + cs,
                   &Kt[r * KTS + cs]);
        }
        // V tile: 96 rows x 64 el = 768 x 16B chunks, 3 per thread
        #pragma unroll
        for (int j = 0; j < 3; ++j) {
            const int c = tid + j * 256;
            const int d = c >> 3, seg = (c & 7) * 8;
            copy16(vtb + (size_t)(b * 768 + h * 96 + d) * 4096 + nc + seg,
                   &Vt[d * VTS + seg]);
        }
        wait_copies();
        __syncthreads();

        // sim = Q @ K^T  (32 queries x 64 keys per wave)
        v8f s[2][4];
        #pragma unroll
        for (int mt = 0; mt < 2; ++mt)
            #pragma unroll
            for (int nt = 0; nt < 4; ++nt) s[mt][nt] = vzero;
        #pragma unroll
        for (int kk = 0; kk < 3; ++kk)
            #pragma unroll
            for (int nt = 0; nt < 4; ++nt) {
                Frag kf;
                const int off = (nt * 16 + l16) * KTS + kk * 32 + bkb;
                kf.q[0] = *(const uint4*)&Kt[off];
                kf.q[1] = *(const uint4*)&Kt[off + 16];
                #pragma unroll
                for (int mt = 0; mt < 2; ++mt)
                    s[mt][nt] = __builtin_amdgcn_wmma_f32_16x16x32_bf16(
                        false, qf[mt][kk].v, false, kf.v, (short)0, s[mt][nt],
                        false, false);
            }

        // online softmax update (row reductions stay in 16-lane halves)
        #pragma unroll
        for (int mt = 0; mt < 2; ++mt) {
            float alpha_r[8];
            #pragma unroll
            for (int r = 0; r < 8; ++r) {
                float mx = fmaxf(fmaxf(s[mt][0][r], s[mt][1][r]),
                                 fmaxf(s[mt][2][r], s[mt][3][r]));
                mx = fmaxf(mx, __shfl_xor(mx, 8));
                mx = fmaxf(mx, __shfl_xor(mx, 4));
                mx = fmaxf(mx, __shfl_xor(mx, 2));
                mx = fmaxf(mx, __shfl_xor(mx, 1));
                const float nm = fmaxf(mrun[mt][r], mx);
                alpha_r[r] = __expf(mrun[mt][r] - nm);
                mrun[mt][r] = nm;
                float rs = 0.f;
                #pragma unroll
                for (int nt = 0; nt < 4; ++nt) {
                    const float p = __expf(s[mt][nt][r] - nm);
                    s[mt][nt][r] = p;
                    rs += p;
                }
                rs += __shfl_xor(rs, 8);
                rs += __shfl_xor(rs, 4);
                rs += __shfl_xor(rs, 2);
                rs += __shfl_xor(rs, 1);
                lrun[mt][r] = lrun[mt][r] * alpha_r[r] + rs;
            }
            #pragma unroll
            for (int dt = 0; dt < 6; ++dt)
                #pragma unroll
                for (int r = 0; r < 8; ++r) oacc[mt][dt][r] *= alpha_r[r];
            // stage P (D-layout) to per-wave LDS for A-fragment reload
            #pragma unroll
            for (int nt = 0; nt < 4; ++nt)
                #pragma unroll
                for (int r = 0; r < 8; ++r)
                    Ps[wave][(mt * 16 + lhi * 8 + r) * PSS + nt * 16 + l16] =
                        f32_to_bf16(s[mt][nt][r]);
        }

        // O += P @ V
        #pragma unroll
        for (int kk = 0; kk < 2; ++kk) {
            Frag pf[2];
            #pragma unroll
            for (int mt = 0; mt < 2; ++mt) {
                const int off = (mt * 16 + l16) * PSS + kk * 32 + akb;
                pf[mt].q[0] = *(const uint4*)&Ps[wave][off];
                pf[mt].q[1] = *(const uint4*)&Ps[wave][off + 16];
            }
            #pragma unroll
            for (int dt = 0; dt < 6; ++dt) {
                Frag vf;
                const int off = (dt * 16 + l16) * VTS + kk * 32 + bkb;
                vf.q[0] = *(const uint4*)&Vt[off];
                vf.q[1] = *(const uint4*)&Vt[off + 16];
                #pragma unroll
                for (int mt = 0; mt < 2; ++mt)
                    oacc[mt][dt] = __builtin_amdgcn_wmma_f32_16x16x32_bf16(
                        false, pf[mt].v, false, vf.v, (short)0, oacc[mt][dt],
                        false, false);
            }
        }
    }

    // normalize and write bf16 output rows
    #pragma unroll
    for (int mt = 0; mt < 2; ++mt)
        #pragma unroll
        for (int dt = 0; dt < 6; ++dt)
            #pragma unroll
            for (int r = 0; r < 8; ++r) {
                const int m = wave * 32 + mt * 16 + lhi * 8 + r;
                const int d = dt * 16 + l16;
                const float val = oacc[mt][dt][r] / lrun[mt][r];
                ob[(size_t)(b * 256 + m) * 768 + h * 96 + d] = f32_to_bf16(val);
            }
}

// ---------------------------------------------------------------------------
// Launch
// ---------------------------------------------------------------------------
extern "C" void kernel_launch(void* const* d_in, const int* in_sizes, int n_in,
                              void* d_out, int out_size, void* d_ws, size_t ws_size,
                              hipStream_t stream) {
    const float* x      = (const float*)d_in[0];   // [8,4096,1024]
    const float* query  = (const float*)d_in[1];   // [256,768]
    const float* ln_k_g = (const float*)d_in[2];
    const float* ln_k_b = (const float*)d_in[3];
    const float* ln_q_g = (const float*)d_in[4];
    const float* ln_q_b = (const float*)d_in[5];
    const float* W_q    = (const float*)d_in[6];   // [768,768]
    const float* W_kv   = (const float*)d_in[7];   // [1024,1536]
    const float* W_out  = (const float*)d_in[8];   // [768,768]
    float* out = (float*)d_out;                    // [8,256,768] fp32

    char* w = (char*)d_ws;
    size_t off = 0;
    auto alloc = [&](size_t bytes) -> u16* {
        u16* p = (u16*)(w + off);
        off += (bytes + 255) & ~(size_t)255;
        return p;
    };
    u16* xk    = alloc((size_t)32768 * 1024 * 2);  // LN(x) bf16
    u16* kbuf  = alloc((size_t)32768 * 768 * 2);   // K rows bf16
    u16* vtbuf = alloc((size_t)8 * 768 * 4096 * 2);// V transposed bf16
    u16* WkvT  = alloc((size_t)1536 * 1024 * 2);   // W_kv^T bf16
    u16* qn    = alloc((size_t)256 * 768 * 2);     // LN(query) bf16
    u16* WqT   = alloc((size_t)768 * 768 * 2);     // W_q^T bf16
    u16* qp    = alloc((size_t)256 * 768 * 2);     // scaled Q bf16
    u16* att   = alloc((size_t)2048 * 768 * 2);    // attention output bf16
    u16* WoT   = alloc((size_t)768 * 768 * 2);     // W_out^T bf16

    // one-time weight prep: convert + transpose so GEMM LDS fills are copies
    cvt_f32_bf16_T_kernel<<<(1024 * 1536 + 255) / 256, 256, 0, stream>>>(W_kv, WkvT, 1024, 1536);
    cvt_f32_bf16_T_kernel<<<(768 * 768 + 255) / 256, 256, 0, stream>>>(W_q, WqT, 768, 768);
    cvt_f32_bf16_T_kernel<<<(768 * 768 + 255) / 256, 256, 0, stream>>>(W_out, WoT, 768, 768);

    // layernorms
    ln_bf16_kernel<<<32768, 256, 0, stream>>>(x, ln_k_g, ln_k_b, xk, 1024);
    ln_bf16_kernel<<<256, 256, 0, stream>>>(query, ln_q_g, ln_q_b, qn, 768);

    // KV projection: [32768,1024] @ [1024,1536]; K half -> rows, V half -> transposed
    gemm_bf16_128x128<2><<<dim3(1536 / 128, 32768 / 128), 256, 0, stream>>>(
        xk, WkvT, kbuf, vtbuf, 32768, 1536, 1024, 1.0f);

    // Q projection with dh^-0.5 folded in: [256,768] @ [768,768] -> bf16
    const float scale = 0.1020620726159658f;  // 96^-0.5
    gemm_bf16_128x128<1><<<dim3(768 / 128, 256 / 128), 256, 0, stream>>>(
        qn, WqT, qp, nullptr, 256, 768, 768, scale);

    // flash attention over 64 (h,b) problems
    flash_attn_kernel<<<dim3(8, 8), 256, 0, stream>>>(qp, kbuf, vtbuf, att);

    // output projection: [2048,768] @ [768,768] -> fp32 d_out
    gemm_bf16_128x128<0><<<dim3(768 / 128, 2048 / 128), 256, 0, stream>>>(
        att, WoT, out, nullptr, 2048, 768, 768, 1.0f);
}
